// MoELayer_39419209842955
// MI455X (gfx1250) — matrix operand order repaired and checked by reading, not compiled, not run on hardware
//
#include <hip/hip_runtime.h>
#include <hip/hip_bf16.h>
#include <math.h>

#define H     1024
#define FFN   4096
#define E_    8
#define T_    8192
#define TOPK  2

#define TM      32          // tokens per tile
#define THREADS 512         // 16 wave32 waves
#define WAVES   16
#define FC      512         // FFN super-chunk (WAVES * 32)

typedef __attribute__((ext_vector_type(16))) __bf16       v16bf;
typedef __attribute__((ext_vector_type(8)))  float        v8f;
typedef __attribute__((ext_vector_type(4)))  unsigned int v4u;

union BfVec { v16bf v; v4u q[2]; };

// ---- optional CDNA5 async global->LDS path (guarded: falls back if absent)
#if defined(__gfx1250__) && \
    __has_builtin(__builtin_amdgcn_global_load_async_to_lds_b128) && \
    __has_builtin(__builtin_amdgcn_s_wait_asynccnt)
#define USE_ASYNC_LDS 1
// signature (from hipcc diagnostic): first param is AS1 pointer to
// GCC-vector int4, non-const; assume matching AS3 pointer for LDS side.
typedef int gv4i __attribute__((vector_size(16)));
typedef __attribute__((address_space(1))) gv4i* g_v4i_ptr;
typedef __attribute__((address_space(3))) gv4i* l_v4i_ptr;
#else
#define USE_ASYNC_LDS 0
#endif

// ---------------------------------------------------------------- zero init
__global__ void zero_init(float* __restrict__ out, size_t n,
                          int* __restrict__ cnt, float* __restrict__ psum) {
    size_t i = (size_t)blockIdx.x * blockDim.x + threadIdx.x;
    size_t stride = (size_t)gridDim.x * blockDim.x;
    for (; i < n; i += stride) out[i] = 0.f;
    if (blockIdx.x == 0 && threadIdx.x < E_) {
        cnt[threadIdx.x]  = 0;
        psum[threadIdx.x] = 0.f;
    }
}

// ------------------------------------------------------- fp32 -> bf16 convert
__global__ void f32_to_bf16(const float* __restrict__ src,
                            __hip_bfloat16* __restrict__ dst, size_t n) {
    size_t i = (size_t)blockIdx.x * blockDim.x + threadIdx.x;
    size_t stride = (size_t)gridDim.x * blockDim.x;
    for (; i < n; i += stride) dst[i] = __float2bfloat16(src[i]);
}

// ---------------------------------------------------------------- router
__global__ __launch_bounds__(256) void router_kernel(
        const float* __restrict__ x, const float* __restrict__ rw,
        int* __restrict__ cnt, float* __restrict__ psum,
        int* __restrict__ etok, float* __restrict__ ewt) {
    __shared__ float rws[E_ * H];
    for (int i = threadIdx.x; i < E_ * H; i += 256) rws[i] = rw[i];
    __syncthreads();

    const int t = blockIdx.x * 256 + threadIdx.x;
    const float* xr = x + (size_t)t * H;

    float acc[E_];
#pragma unroll
    for (int e = 0; e < E_; ++e) acc[e] = 0.f;

    for (int h = 0; h < H; ++h) {
        float xv = xr[h];
#pragma unroll
        for (int e = 0; e < E_; ++e) acc[e] += xv * rws[e * H + h];
    }

    float m = acc[0];
#pragma unroll
    for (int e = 1; e < E_; ++e) m = fmaxf(m, acc[e]);
    float p[E_], s = 0.f;
#pragma unroll
    for (int e = 0; e < E_; ++e) { p[e] = expf(acc[e] - m); s += p[e]; }
    float inv = 1.f / s;
#pragma unroll
    for (int e = 0; e < E_; ++e) p[e] *= inv;

    int i0 = 0; float w0 = p[0];
#pragma unroll
    for (int e = 1; e < E_; ++e) if (p[e] > w0) { w0 = p[e]; i0 = e; }
    int i1 = -1; float w1 = -1.f;
#pragma unroll
    for (int e = 0; e < E_; ++e) if (e != i0 && p[e] > w1) { w1 = p[e]; i1 = e; }
    float norm = 1.f / (w0 + w1);
    w0 *= norm; w1 *= norm;

#pragma unroll
    for (int e = 0; e < E_; ++e) atomicAdd(&psum[e], p[e]);

    int s0 = atomicAdd(&cnt[i0], 1);
    etok[i0 * T_ + s0] = t;  ewt[i0 * T_ + s0] = w0;
    int s1 = atomicAdd(&cnt[i1], 1);
    etok[i1 * T_ + s1] = t;  ewt[i1 * T_ + s1] = w1;
}

// --------------------------------------------------------- fused expert FFN
// Block = 512 threads = 16 wave32 waves; tile = 32 tokens x full H output.
// Stage 1 per super-chunk: wave w computes h[:, w*32 .. w*32+32) with K=H.
// Stage 2: wave w owns out columns [w*64, w*64+64), K = FC per super-chunk.
__global__ __launch_bounds__(THREADS, 1) void expert_ffn(
        const __hip_bfloat16* __restrict__ w1b, const __hip_bfloat16* __restrict__ w2b,
        const float* __restrict__ b1, const float* __restrict__ b2,
        const float* __restrict__ x, const int* __restrict__ cnt,
        const int* __restrict__ etok, const float* __restrict__ ewt,
        float* __restrict__ out) {
    const int e   = blockIdx.y;
    const int n_e = cnt[e];
    const int t0  = blockIdx.x * TM;
    if (t0 >= n_e) return;

    __shared__ __hip_bfloat16 xs[TM][H + 8];       // token tile, bf16 (~64.5KB)
    __shared__ __hip_bfloat16 hs[TM][FC + 16];     // gelu(h) super-chunk (~33KB)
    __shared__ int   toks[TM];
    __shared__ float wts[TM];

    const int tid = threadIdx.x;
    if (tid < TM) {
        int idx  = t0 + tid;
        bool vld = idx < n_e;
        toks[tid] = etok[e * T_ + (vld ? idx : t0)];
        wts[tid]  = vld ? ewt[e * T_ + idx] : 0.f;   // padded rows contribute 0
    }
    __syncthreads();

#if USE_ASYNC_LDS
    // stage x tile via async global->LDS (f32 chunks through hs buffer),
    // then LDS-side convert to bf16. hs row = 528 bf16 = 1056B >= 256 f32.
    {
        float* hsf = reinterpret_cast<float*>(&hs[0][0]);
        const int RSF = (FC + 16) / 2;                 // 264 floats per row
        for (int c4 = 0; c4 < H / 256; ++c4) {
#pragma unroll
            for (int p = 0; p < 4; ++p) {              // 2048 16B packets / 512 thr
                int pk = p * THREADS + tid;
                int m  = pk >> 6;                      // row 0..31
                int g  = pk & 63;                      // 16B group in row chunk
                const float* gp = x + (size_t)toks[m] * H + c4 * 256 + g * 4;
                float*       lp = hsf + m * RSF + g * 4;
                __builtin_amdgcn_global_load_async_to_lds_b128(
                    (g_v4i_ptr)(uintptr_t)gp,
                    (l_v4i_ptr)(uintptr_t)lp,
                    0, 0);
            }
            __builtin_amdgcn_s_wait_asynccnt(0);
            __syncthreads();
            for (int i = tid; i < TM * 256; i += THREADS) {
                int m = i >> 8, hh = i & 255;
                xs[m][c4 * 256 + hh] = __float2bfloat16(hsf[m * RSF + hh]);
            }
            __syncthreads();
        }
    }
#else
    for (int i = tid; i < TM * H; i += THREADS) {
        int m = i >> 10, hh = i & (H - 1);
        xs[m][hh] = __float2bfloat16(x[(size_t)toks[m] * H + hh]);
    }
    __syncthreads();
#endif

    const int wave  = tid >> 5;
    const int lane  = tid & 31;
    const int lrow  = lane & 15;   // A row / B-C-D column
    const int lhalf = lane >> 4;   // K/M half select

    const __hip_bfloat16* W1 = w1b + (size_t)e * FFN * H;
    const __hip_bfloat16* W2 = w2b + (size_t)e * H * FFN;
    __builtin_prefetch(W1, 0, 1);
    __builtin_prefetch(W2, 0, 1);

    const v8f z = {0.f, 0.f, 0.f, 0.f, 0.f, 0.f, 0.f, 0.f};
    v8f y[2][4];                       // [m-tile][n-tile]
#pragma unroll
    for (int mt = 0; mt < 2; ++mt)
#pragma unroll
        for (int nt = 0; nt < 4; ++nt) y[mt][nt] = z;

    for (int cs = 0; cs < FFN / FC; ++cs) {
        // ---- stage 1: h[:, fb..fb+32) = gelu(x @ W1^T + b1), K = H
        const int fb = cs * FC + wave * 32;
        v8f ha[2][2];                  // [m-tile][h n-tile]
#pragma unroll
        for (int mt = 0; mt < 2; ++mt) { ha[mt][0] = z; ha[mt][1] = z; }

        for (int k = 0; k < H / 32; ++k) {
            const int kb = k * 32;
            BfVec a0, a1;              // A rows: lrow and 16+lrow
            a0.q[0] = *(const v4u*)&xs[lrow][kb + lhalf * 8];
            a0.q[1] = *(const v4u*)&xs[lrow][kb + 16 + lhalf * 8];
            a1.q[0] = *(const v4u*)&xs[16 + lrow][kb + lhalf * 8];
            a1.q[1] = *(const v4u*)&xs[16 + lrow][kb + 16 + lhalf * 8];
            BfVec b0, b1v;             // B columns: fb+lrow and fb+16+lrow
            const __hip_bfloat16* p0 = W1 + (size_t)(fb + lrow) * H + kb + lhalf * 8;
            const __hip_bfloat16* p1 = W1 + (size_t)(fb + 16 + lrow) * H + kb + lhalf * 8;
            b0.q[0]  = *(const v4u*)p0;  b0.q[1]  = *(const v4u*)(p0 + 16);
            b1v.q[0] = *(const v4u*)p1;  b1v.q[1] = *(const v4u*)(p1 + 16);
            ha[0][0] = __builtin_amdgcn_wmma_f32_16x16x32_bf16(false, a0.v, false, b0.v,
                                                               (short)0, ha[0][0], false, false);
            ha[0][1] = __builtin_amdgcn_wmma_f32_16x16x32_bf16(false, a0.v, false, b1v.v,
                                                               (short)0, ha[0][1], false, false);
            ha[1][0] = __builtin_amdgcn_wmma_f32_16x16x32_bf16(false, a1.v, false, b0.v,
                                                               (short)0, ha[1][0], false, false);
            ha[1][1] = __builtin_amdgcn_wmma_f32_16x16x32_bf16(false, a1.v, false, b1v.v,
                                                               (short)0, ha[1][1], false, false);
        }
        {   // bias + exact gelu + bf16 store (C layout: row = j + 8*lhalf)
            float bn0 = b1[e * FFN + fb + lrow];
            float bn1 = b1[e * FFN + fb + 16 + lrow];
#pragma unroll
            for (int mt = 0; mt < 2; ++mt) {
#pragma unroll
                for (int j = 0; j < 8; ++j) {
                    int m = mt * 16 + j + 8 * lhalf;
                    float v0 = ha[mt][0][j] + bn0;
                    float v1 = ha[mt][1][j] + bn1;
                    v0 = 0.5f * v0 * (1.f + erff(v0 * 0.70710678118f));
                    v1 = 0.5f * v1 * (1.f + erff(v1 * 0.70710678118f));
                    hs[m][wave * 32 + lrow]      = __float2bfloat16(v0);
                    hs[m][wave * 32 + 16 + lrow] = __float2bfloat16(v1);
                }
            }
        }
        __syncthreads();

        // ---- stage 2: y[:, wave*64 .. +64) += h_chunk @ W2_chunk^T, K = FC
        for (int k = 0; k < FC / 32; ++k) {
            const int kb = k * 32;
            BfVec a0, a1;
            a0.q[0] = *(const v4u*)&hs[lrow][kb + lhalf * 8];
            a0.q[1] = *(const v4u*)&hs[lrow][kb + 16 + lhalf * 8];
            a1.q[0] = *(const v4u*)&hs[16 + lrow][kb + lhalf * 8];
            a1.q[1] = *(const v4u*)&hs[16 + lrow][kb + 16 + lhalf * 8];
            const int fbase = cs * FC + kb + lhalf * 8;
#pragma unroll
            for (int nt = 0; nt < 4; ++nt) {
                const int n = wave * 64 + nt * 16 + lrow;
                BfVec b;
                const __hip_bfloat16* p = W2 + (size_t)n * FFN + fbase;
                b.q[0] = *(const v4u*)p;  b.q[1] = *(const v4u*)(p + 16);
                y[0][nt] = __builtin_amdgcn_wmma_f32_16x16x32_bf16(false, a0.v, false, b.v,
                                                                   (short)0, y[0][nt], false, false);
                y[1][nt] = __builtin_amdgcn_wmma_f32_16x16x32_bf16(false, a1.v, false, b.v,
                                                                   (short)0, y[1][nt], false, false);
            }
        }
        __syncthreads();
    }

    // ---- combine: out[token] += wt * (y + b2)
#pragma unroll
    for (int mt = 0; mt < 2; ++mt) {
#pragma unroll
        for (int nt = 0; nt < 4; ++nt) {
            int n = wave * 64 + nt * 16 + lrow;
            float bias = b2[e * H + n];
#pragma unroll
            for (int j = 0; j < 8; ++j) {
                int m = mt * 16 + j + 8 * lhalf;
                float val = (y[mt][nt][j] + bias) * wts[m];
                atomicAdd(&out[(size_t)toks[m] * H + n], val);
            }
        }
    }
}

// ------------------------------------------------------------ aux-loss finish
__global__ void finalize_aux(const int* __restrict__ cnt,
                             const float* __restrict__ psum,
                             float* __restrict__ aux) {
    if (blockIdx.x == 0 && threadIdx.x == 0) {
        float s = 0.f;
        for (int e = 0; e < E_; ++e) {
            float freq = (float)cnt[e] / (float)(T_ * TOPK);
            float mp   = psum[e] / (float)T_;
            s += mp * freq;
        }
        aux[0] = 0.01f * (float)E_ * s;
    }
}

// ---------------------------------------------------------------- launcher
extern "C" void kernel_launch(void* const* d_in, const int* in_sizes, int n_in,
                              void* d_out, int out_size, void* d_ws, size_t ws_size,
                              hipStream_t stream) {
    const float* x    = (const float*)d_in[0];
    const float* rw   = (const float*)d_in[1];
    const float* fc1w = (const float*)d_in[2];
    const float* fc1b = (const float*)d_in[3];
    const float* fc2w = (const float*)d_in[4];
    const float* fc2b = (const float*)d_in[5];
    float* out = (float*)d_out;

    char* ws = (char*)d_ws;
    size_t off = 0;
    __hip_bfloat16* w1b = (__hip_bfloat16*)(ws + off); off += (size_t)E_ * FFN * H * 2;
    __hip_bfloat16* w2b = (__hip_bfloat16*)(ws + off); off += (size_t)E_ * H * FFN * 2;
    int*   etok = (int*)  (ws + off); off += (size_t)E_ * T_ * 4;
    float* ewt  = (float*)(ws + off); off += (size_t)E_ * T_ * 4;
    int*   cnt  = (int*)  (ws + off); off += 64;
    float* psum = (float*)(ws + off); off += 64;

    zero_init<<<2048, 256, 0, stream>>>(out, (size_t)T_ * H + 1, cnt, psum);

    const size_t nw = (size_t)E_ * FFN * H;
    f32_to_bf16<<<4096, 256, 0, stream>>>(fc1w, w1b, nw);
    f32_to_bf16<<<4096, 256, 0, stream>>>(fc2w, w2b, nw);

    router_kernel<<<T_ / 256, 256, 0, stream>>>(x, rw, cnt, psum, etok, ewt);

    dim3 grid(T_ / TM, E_);
    expert_ffn<<<grid, THREADS, 0, stream>>>(w1b, w2b, fc1b, fc2b, x, cnt, etok, ewt, out);

    finalize_aux<<<1, 32, 0, stream>>>(cnt, psum, out + (size_t)T_ * H);
}
